// SEGNN_73031623901572
// MI455X (gfx1250) — matrix operand order repaired and checked by reading, not compiled, not run on hardware
//
#include <hip/hip_runtime.h>
#include <hip/hip_bf16.h>

// ---------------------------------------------------------------------------
// SEGNN on MI455X (gfx1250): all dense o3_tp matmuls run on
// v_wmma_f32_16x16x32_f16 (16 rows x K(32-chunks) x 16 cols, f32 accum).
// Weights are pre-packed into WMMA B-fragment layout (f16) in workspace.
// One wave32 per block; A operands staged in LDS row-major f16 so each lane
// fetches its fragment with two 16B ds loads matching the ISA A layout.
// ---------------------------------------------------------------------------

typedef __attribute__((ext_vector_type(16))) _Float16 v16h;
typedef __attribute__((ext_vector_type(8)))  _Float16 v8h;
typedef __attribute__((ext_vector_type(8)))  float    v8f;

#define DEVINL __device__ __forceinline__

DEVINL v8f vz() {
    v8f z;
#pragma unroll
    for (int j = 0; j < 8; ++j) z[j] = 0.f;
    return z;
}

// A fragment (16x32 f16) from row-major LDS buffer, per ISA 16-bit A layout:
// lane l: row m = l&15; halves 0..7 -> K = (l>>4)*8 + 0..7,
//                       halves 8..15 -> K = 16 + (l>>4)*8 + 0..7.
DEVINL v16h load_a_frag(const _Float16* A, int lda, int col0, int lane) {
    const int m  = lane & 15;
    const int kh = (lane >> 4) * 8;
    const _Float16* p = A + m * lda + col0 + kh;
    union { v16h v; v8h h[2]; } u;
    u.h[0] = *(const v8h*)(p);
    u.h[1] = *(const v8h*)(p + 16);
    return u.v;
}

// C += A(16xCH*32) @ B(K x NT*16), B pre-packed: frag fi=(c*NT+t), lane l,
// halves are 32 contiguous bytes at B + (fi*32+l)*16.
template<int NT, int CH>
DEVINL void gemm_nt(const _Float16* A, int lda, int col0,
                    const _Float16* __restrict__ B, v8f* acc, int lane) {
#pragma unroll
    for (int c = 0; c < CH; ++c) {
        v16h af = load_a_frag(A, lda, col0 + c * 32, lane);
#pragma unroll
        for (int t = 0; t < NT; ++t) {
            v16h bf = *(const v16h*)(B + (size_t)((c * NT + t) * 32 + lane) * 16);
            acc[t] = __builtin_amdgcn_wmma_f32_16x16x32_f16(
                false, af, false, bf, (short)0, acc[t], false, false);
        }
    }
}

DEVINL float sigm(float x) { return 1.f / (1.f + __expf(-x)); }

// ---------------------------------------------------------------------------
// Weight packing: X @ W^T  =>  B[k][n] = W[n][k], W row-major (O x K) fp32.
// grid = chunks*NT blocks of 32 threads; each lane writes its 16 halves.
// ---------------------------------------------------------------------------
__global__ __launch_bounds__(32) void pack_b_kernel(
    const float* __restrict__ W, int O, int K, int NT, _Float16* __restrict__ dst) {
    int fi = blockIdx.x;
    int t = fi % NT;
    int c = fi / NT;
    int lane = threadIdx.x;
    int n  = t * 16 + (lane & 15);
    int kb = c * 32 + (lane >> 4) * 8;
    _Float16* d = dst + ((size_t)fi * 32 + lane) * 16;
#pragma unroll
    for (int h = 0; h < 16; ++h) {
        int k = kb + (h < 8 ? h : h + 8);
        float v = (n < O && k < K) ? W[(size_t)n * K + k] : 0.f;
        d[h] = (_Float16)v;
    }
}

// ---------------------------------------------------------------------------
// Embedding (tiny: m0=1, m1=2) -- plain VALU kernel. na_s == 1.
// ---------------------------------------------------------------------------
__global__ void embed_kernel(
    const float* __restrict__ x, const float* __restrict__ nattr,
    const float* __restrict__ Wss, const float* __restrict__ Wsv,
    const float* __restrict__ Wvs, const float* __restrict__ Wvv,
    const float* __restrict__ b,
    float* __restrict__ hs, float* __restrict__ hv, int N) {
    int n = blockIdx.x * blockDim.x + threadIdx.x;
    if (n >= N) return;
    const float c0 = 0.5773502691896258f; // 1/sqrt(3)
    float xs = x[(size_t)n * 7 + 6];
    float xv[2][3];
#pragma unroll
    for (int i = 0; i < 2; ++i)
#pragma unroll
        for (int c = 0; c < 3; ++c) xv[i][c] = x[(size_t)n * 7 + i * 3 + c];
    float nav[3] = { nattr[(size_t)n * 4 + 1], nattr[(size_t)n * 4 + 2],
                     nattr[(size_t)n * 4 + 3] };
    float dot[2];
#pragma unroll
    for (int i = 0; i < 2; ++i)
        dot[i] = (xv[i][0] * nav[0] + xv[i][1] * nav[1] + xv[i][2] * nav[2]) * c0;
#pragma unroll
    for (int o = 0; o < 32; ++o) {
        hs[(size_t)n * 32 + o] =
            (Wss[o] * xs + Wvv[o * 2] * dot[0] + Wvv[o * 2 + 1] * dot[1]) * c0 + b[o];
#pragma unroll
        for (int c = 0; c < 3; ++c)
            hv[(size_t)n * 96 + o * 3 + c] =
                ((Wvs[o * 2] * xv[0][c] + Wvs[o * 2 + 1] * xv[1][c]) +
                 Wsv[o] * xs * nav[c]) * c0;
    }
}

// ---------------------------------------------------------------------------
// Edge message kernel: gather -> msg1(WMMA)+gate -> msg2(WMMA)+gate -> scatter
// ---------------------------------------------------------------------------
struct EdgeW {
    const _Float16 *ss1, *vv1, *sv1, *vs1; const float* b1;
    const _Float16 *ss2, *vv2, *sv2, *vs2; const float* b2;
};

__global__ __launch_bounds__(32) void edge_msg_kernel(
    const float* __restrict__ hs, const float* __restrict__ hv,
    const int* __restrict__ eidx, const float* __restrict__ eattr,
    const float* __restrict__ amf,
    float* __restrict__ ag_s, float* __restrict__ ag_v,
    EdgeW w, int E) {
    // A = [ms(66) | pad->96 | dot(64)] : K = 160
    __shared__ alignas(32) _Float16 Abuf[16][160];
    __shared__ alignas(32) _Float16 Vbuf[3][16][64];
    __shared__ float Sbuf[16][64];
    __shared__ float Vout[16][32][3];
    __shared__ int   sidx_s[16], didx_s[16];
    __shared__ float eas[16];
    __shared__ float eav[16][3];

    const int lane  = threadIdx.x;
    const int r     = lane & 15;
    const int half  = lane >> 4;
    const int mbase = half * 8;
    const int ntile = (E + 15) >> 4;
    const float inv_sqrt3 = 0.5773502691896258f;
    const float c1 = 0.08770580193070292f; // 1/sqrt(130)
    const float c2 = 0.125f;               // 1/sqrt(64)

    for (int tile = blockIdx.x; tile < ntile; tile += gridDim.x) {
        // phase 0: per-edge attrs / indices
        {
            int e  = tile * 16 + r;
            bool v = (e < E);
            int ec = v ? e : (E - 1);
            if (half == 0) {
                sidx_s[r] = eidx[ec];
                didx_s[r] = eidx[E + ec];
                eas[r]    = v ? eattr[(size_t)ec * 4 + 0] : 0.f;
                eav[r][0] = v ? eattr[(size_t)ec * 4 + 1] : 0.f;
                eav[r][1] = v ? eattr[(size_t)ec * 4 + 2] : 0.f;
                eav[r][2] = v ? eattr[(size_t)ec * 4 + 3] : 0.f;
                Abuf[r][64] = (_Float16)(v ? amf[(size_t)ec * 2 + 0] : 0.f);
                Abuf[r][65] = (_Float16)(v ? amf[(size_t)ec * 2 + 1] : 0.f);
#pragma unroll
                for (int j = 66; j < 96; ++j) Abuf[r][j] = (_Float16)0.f;
            }
        }
        __syncthreads();
        // phase 1: gather hs[dst],hs[src] and hv[dst],hv[src]
        {
            int node = half ? sidx_s[r] : didx_s[r];
            const float* hsr = hs + (size_t)node * 32;
#pragma unroll
            for (int j = 0; j < 32; ++j)
                Abuf[r][half * 32 + j] = (_Float16)hsr[j];
            const float* hvr = hv + (size_t)node * 96;
#pragma unroll
            for (int q = 0; q < 96; ++q)
                Vbuf[q % 3][r][half * 32 + q / 3] = (_Float16)hvr[q];
        }
        __syncthreads();
        // phase 2: dot = mv . ea_v / sqrt(3) -> A cols 96..159
        {
            float a0 = eav[r][0], a1 = eav[r][1], a2 = eav[r][2];
#pragma unroll
            for (int j = 0; j < 32; ++j) {
                int i = half * 32 + j;
                float d = ((float)Vbuf[0][r][i] * a0 + (float)Vbuf[1][r][i] * a1 +
                           (float)Vbuf[2][r][i] * a2) * inv_sqrt3;
                Abuf[r][96 + i] = (_Float16)d;
            }
        }
        __syncthreads();
        // phase 3: msg1 GEMMs + combine
        {
            v8f aS1[4] = { vz(), vz(), vz(), vz() };
            v8f aS2[4] = { vz(), vz(), vz(), vz() };
            gemm_nt<4, 3>(&Abuf[0][0], 160, 0,  w.ss1, aS1, lane);
            gemm_nt<4, 2>(&Abuf[0][0], 160, 96, w.vv1, aS2, lane);
            v8f aT[2] = { vz(), vz() };
            gemm_nt<2, 3>(&Abuf[0][0], 160, 0,  w.sv1, aT, lane);
#pragma unroll
            for (int t = 0; t < 4; ++t) {
                int n = t * 16 + r;
                float bb = w.b1[n];
#pragma unroll
                for (int q = 0; q < 8; ++q) {
                    int m = mbase + q;
                    Sbuf[m][n] = (eas[m] * aS1[t][q] + aS2[t][q]) * c1 + bb;
                }
            }
#pragma unroll
            for (int c = 0; c < 3; ++c) {
                v8f aV[2] = { vz(), vz() };
                gemm_nt<2, 2>(&Vbuf[c][0][0], 64, 0, w.vs1, aV, lane);
#pragma unroll
                for (int t = 0; t < 2; ++t) {
                    int n = t * 16 + r;
#pragma unroll
                    for (int q = 0; q < 8; ++q) {
                        int m = mbase + q;
                        Vout[m][n][c] = (eas[m] * aV[t][q] + aT[t][q] * eav[m][c]) * c1;
                    }
                }
            }
        }
        __syncthreads();
        // phase 4: gate1 -> ms2 (A cols 0..31), mv2 (Vbuf cols 0..31), dot2 (A 32..63)
        {
#pragma unroll
            for (int j = 0; j < 16; ++j) {
                int o = half * 16 + j;
                float s = Sbuf[r][o];
                Abuf[r][o] = (_Float16)(s * sigm(s));
                float g = sigm(Sbuf[r][32 + o]);
#pragma unroll
                for (int c = 0; c < 3; ++c)
                    Vbuf[c][r][o] = (_Float16)(g * Vout[r][o][c]);
            }
#pragma unroll
            for (int j = 0; j < 16; ++j) {
                int i = half * 16 + j;
                float d = ((float)Vbuf[0][r][i] * eav[r][0] +
                           (float)Vbuf[1][r][i] * eav[r][1] +
                           (float)Vbuf[2][r][i] * eav[r][2]) * inv_sqrt3;
                Abuf[r][32 + i] = (_Float16)d;
            }
        }
        __syncthreads();
        // phase 5: msg2 GEMMs + combine
        {
            v8f bS1[4] = { vz(), vz(), vz(), vz() };
            v8f bS2[4] = { vz(), vz(), vz(), vz() };
            gemm_nt<4, 1>(&Abuf[0][0], 160, 0,  w.ss2, bS1, lane);
            gemm_nt<4, 1>(&Abuf[0][0], 160, 32, w.vv2, bS2, lane);
            v8f bT[2] = { vz(), vz() };
            gemm_nt<2, 1>(&Abuf[0][0], 160, 0,  w.sv2, bT, lane);
#pragma unroll
            for (int t = 0; t < 4; ++t) {
                int n = t * 16 + r;
                float bb = w.b2[n];
#pragma unroll
                for (int q = 0; q < 8; ++q) {
                    int m = mbase + q;
                    Sbuf[m][n] = (eas[m] * bS1[t][q] + bS2[t][q]) * c2 + bb;
                }
            }
#pragma unroll
            for (int c = 0; c < 3; ++c) {
                v8f bV[2] = { vz(), vz() };
                gemm_nt<2, 1>(&Vbuf[c][0][0], 64, 0, w.vs2, bV, lane);
#pragma unroll
                for (int t = 0; t < 2; ++t) {
                    int n = t * 16 + r;
#pragma unroll
                    for (int q = 0; q < 8; ++q) {
                        int m = mbase + q;
                        Vout[m][n][c] = (eas[m] * bV[t][q] + bT[t][q] * eav[m][c]) * c2;
                    }
                }
            }
        }
        __syncthreads();
        // phase 6: gate2 + segment-sum scatter (atomics)
        {
            int e = tile * 16 + r;
            if (e < E) {
                int d = didx_s[r];
                float* ags = ag_s + (size_t)d * 32;
                float* agv = ag_v + (size_t)d * 96;
#pragma unroll
                for (int j = 0; j < 16; ++j) {
                    int o = half * 16 + j;
                    float s = Sbuf[r][o];
                    atomicAdd(&ags[o], s * sigm(s));
                    float g = sigm(Sbuf[r][32 + o]);
#pragma unroll
                    for (int c = 0; c < 3; ++c)
                        atomicAdd(&agv[o * 3 + c], g * Vout[r][o][c]);
                }
            }
        }
        __syncthreads();
    }
}

// ---------------------------------------------------------------------------
// Node update: upd1(WMMA)+gate -> upd2(WMMA) -> residual.  na_s == 1.
// ---------------------------------------------------------------------------
struct NodeW {
    const _Float16 *ss1, *vv1, *sv1, *vs1; const float* b1;
    const _Float16 *ss2, *vv2, *sv2, *vs2; const float* b2;
};

__global__ __launch_bounds__(32) void node_upd_kernel(
    float* __restrict__ hs, float* __restrict__ hv,
    const float* __restrict__ ag_s, const float* __restrict__ ag_v,
    const float* __restrict__ nattr, NodeW w, int N) {
    // A = [hs(32) | ag_s(32) | dot(64)] : K = 128
    __shared__ alignas(32) _Float16 Abuf[16][128];
    __shared__ alignas(32) _Float16 Vbuf[3][16][64];
    __shared__ float Sbuf[16][64];
    __shared__ float Vout[16][32][3];
    __shared__ float nav[16][3];

    const int lane  = threadIdx.x;
    const int r     = lane & 15;
    const int half  = lane >> 4;
    const int mbase = half * 8;
    const int ntile = (N + 15) >> 4;
    const float inv_sqrt3 = 0.5773502691896258f;
    const float cu1 = 0.08838834764831845f; // 1/sqrt(128)
    const float cu2 = 0.125f;               // 1/sqrt(64)

    for (int tile = blockIdx.x; tile < ntile; tile += gridDim.x) {
        int nd = tile * 16 + r;
        int nc = nd < N ? nd : N - 1;
        if (half == 0) {
            nav[r][0] = nattr[(size_t)nc * 4 + 1];
            nav[r][1] = nattr[(size_t)nc * 4 + 2];
            nav[r][2] = nattr[(size_t)nc * 4 + 3];
        }
        // stage us = [hs | ag_s], uv = [hv | ag_v]
        {
            const float* s32 = half ? (ag_s + (size_t)nc * 32) : (hs + (size_t)nc * 32);
#pragma unroll
            for (int j = 0; j < 32; ++j)
                Abuf[r][half * 32 + j] = (_Float16)s32[j];
            const float* v96 = half ? (ag_v + (size_t)nc * 96) : (hv + (size_t)nc * 96);
#pragma unroll
            for (int q = 0; q < 96; ++q)
                Vbuf[q % 3][r][half * 32 + q / 3] = (_Float16)v96[q];
        }
        __syncthreads();
        // dot -> A cols 64..127
        {
            float a0 = nav[r][0], a1 = nav[r][1], a2 = nav[r][2];
#pragma unroll
            for (int j = 0; j < 32; ++j) {
                int i = half * 32 + j;
                float d = ((float)Vbuf[0][r][i] * a0 + (float)Vbuf[1][r][i] * a1 +
                           (float)Vbuf[2][r][i] * a2) * inv_sqrt3;
                Abuf[r][64 + i] = (_Float16)d;
            }
        }
        __syncthreads();
        // upd1 (a_s == 1 -> single accumulator)
        {
            v8f aS[4] = { vz(), vz(), vz(), vz() };
            gemm_nt<4, 2>(&Abuf[0][0], 128, 0,  w.ss1, aS, lane);
            gemm_nt<4, 2>(&Abuf[0][0], 128, 64, w.vv1, aS, lane);
            v8f aT[2] = { vz(), vz() };
            gemm_nt<2, 2>(&Abuf[0][0], 128, 0,  w.sv1, aT, lane);
#pragma unroll
            for (int t = 0; t < 4; ++t) {
                int n = t * 16 + r;
                float bb = w.b1[n];
#pragma unroll
                for (int q = 0; q < 8; ++q) {
                    int m = mbase + q;
                    Sbuf[m][n] = aS[t][q] * cu1 + bb;
                }
            }
#pragma unroll
            for (int c = 0; c < 3; ++c) {
                v8f aV[2] = { vz(), vz() };
                gemm_nt<2, 2>(&Vbuf[c][0][0], 64, 0, w.vs1, aV, lane);
#pragma unroll
                for (int t = 0; t < 2; ++t) {
                    int n = t * 16 + r;
#pragma unroll
                    for (int q = 0; q < 8; ++q) {
                        int m = mbase + q;
                        Vout[m][n][c] = (aV[t][q] + aT[t][q] * nav[m][c]) * cu1;
                    }
                }
            }
        }
        __syncthreads();
        // gate -> A cols 0..31 (silu), Vbuf cols 0..31 (gated), dot2 -> A 32..63
        {
#pragma unroll
            for (int j = 0; j < 16; ++j) {
                int o = half * 16 + j;
                float s = Sbuf[r][o];
                Abuf[r][o] = (_Float16)(s * sigm(s));
                float g = sigm(Sbuf[r][32 + o]);
#pragma unroll
                for (int c = 0; c < 3; ++c)
                    Vbuf[c][r][o] = (_Float16)(g * Vout[r][o][c]);
            }
#pragma unroll
            for (int j = 0; j < 16; ++j) {
                int i = half * 16 + j;
                float d = ((float)Vbuf[0][r][i] * nav[r][0] +
                           (float)Vbuf[1][r][i] * nav[r][1] +
                           (float)Vbuf[2][r][i] * nav[r][2]) * inv_sqrt3;
                Abuf[r][32 + i] = (_Float16)d;
            }
        }
        __syncthreads();
        // upd2 + residual (direct global RMW, each (node,ch) owned by one lane)
        {
            v8f cS[2] = { vz(), vz() };
            gemm_nt<2, 1>(&Abuf[0][0], 128, 0,  w.ss2, cS, lane);
            gemm_nt<2, 1>(&Abuf[0][0], 128, 32, w.vv2, cS, lane);
            v8f cT[2] = { vz(), vz() };
            gemm_nt<2, 1>(&Abuf[0][0], 128, 0,  w.sv2, cT, lane);
#pragma unroll
            for (int t = 0; t < 2; ++t) {
                int n = t * 16 + r;
                float bb = w.b2[n];
#pragma unroll
                for (int q = 0; q < 8; ++q) {
                    int m = mbase + q;
                    int node = tile * 16 + m;
                    if (node < N)
                        hs[(size_t)node * 32 + n] += cS[t][q] * cu2 + bb;
                }
            }
#pragma unroll
            for (int c = 0; c < 3; ++c) {
                v8f cV[2] = { vz(), vz() };
                gemm_nt<2, 1>(&Vbuf[c][0][0], 64, 0, w.vs2, cV, lane);
#pragma unroll
                for (int t = 0; t < 2; ++t) {
                    int n = t * 16 + r;
#pragma unroll
                    for (int q = 0; q < 8; ++q) {
                        int m = mbase + q;
                        int node = tile * 16 + m;
                        if (node < N)
                            hv[(size_t)node * 96 + n * 3 + c] +=
                                (cV[t][q] + cT[t][q] * nav[m][c]) * cu2;
                    }
                }
            }
        }
        __syncthreads();
    }
}

// ---------------------------------------------------------------------------
// Final: pre1 gate (WMMA) then pre2 (o1=2, tiny VALU reduction) -> out (N,6)
// ---------------------------------------------------------------------------
struct FinW {
    const _Float16 *ss1, *vv1, *sv1, *vs1; const float* b1;
    const float *vs2, *sv2;
};

__global__ __launch_bounds__(32) void final_kernel(
    const float* __restrict__ hs, const float* __restrict__ hv,
    const float* __restrict__ nattr, FinW w, float* __restrict__ out, int N) {
    __shared__ alignas(32) _Float16 Abuf[16][64];
    __shared__ alignas(32) _Float16 Vbuf[3][16][32];
    __shared__ float Sbuf[16][64];
    __shared__ float Vout[16][32][3];
    __shared__ float Hs2[16][32];
    __shared__ float nav[16][3];

    const int lane  = threadIdx.x;
    const int r     = lane & 15;
    const int half  = lane >> 4;
    const int mbase = half * 8;
    const int ntile = (N + 15) >> 4;
    const float inv_sqrt3 = 0.5773502691896258f;
    const float cp = 0.125f; // 1/sqrt(64)

    for (int tile = blockIdx.x; tile < ntile; tile += gridDim.x) {
        int nd = tile * 16 + r;
        int nc = nd < N ? nd : N - 1;
        if (half == 0) {
            nav[r][0] = nattr[(size_t)nc * 4 + 1];
            nav[r][1] = nattr[(size_t)nc * 4 + 2];
            nav[r][2] = nattr[(size_t)nc * 4 + 3];
        }
        {
            const float* hsr = hs + (size_t)nc * 32;
#pragma unroll
            for (int j = 0; j < 16; ++j)
                Abuf[r][half * 16 + j] = (_Float16)hsr[half * 16 + j];
            const float* hvr = hv + (size_t)nc * 96;
#pragma unroll
            for (int q = 0; q < 48; ++q) {
                int j = half * 16 + q / 3;
                int c = q % 3;
                Vbuf[c][r][j] = (_Float16)hvr[j * 3 + c];
            }
        }
        __syncthreads();
        {
#pragma unroll
            for (int j = 0; j < 16; ++j) {
                int i = half * 16 + j;
                float d = ((float)Vbuf[0][r][i] * nav[r][0] +
                           (float)Vbuf[1][r][i] * nav[r][1] +
                           (float)Vbuf[2][r][i] * nav[r][2]) * inv_sqrt3;
                Abuf[r][32 + i] = (_Float16)d;
            }
        }
        __syncthreads();
        {
            v8f aS[4] = { vz(), vz(), vz(), vz() };
            gemm_nt<4, 1>(&Abuf[0][0], 64, 0,  w.ss1, aS, lane);
            gemm_nt<4, 1>(&Abuf[0][0], 64, 32, w.vv1, aS, lane);
            v8f aT[2] = { vz(), vz() };
            gemm_nt<2, 1>(&Abuf[0][0], 64, 0,  w.sv1, aT, lane);
#pragma unroll
            for (int t = 0; t < 4; ++t) {
                int n = t * 16 + r;
                float bb = w.b1[n];
#pragma unroll
                for (int q = 0; q < 8; ++q) {
                    int m = mbase + q;
                    Sbuf[m][n] = aS[t][q] * cp + bb;
                }
            }
#pragma unroll
            for (int c = 0; c < 3; ++c) {
                v8f aV[2] = { vz(), vz() };
                gemm_nt<2, 1>(&Vbuf[c][0][0], 32, 0, w.vs1, aV, lane);
#pragma unroll
                for (int t = 0; t < 2; ++t) {
                    int n = t * 16 + r;
#pragma unroll
                    for (int q = 0; q < 8; ++q) {
                        int m = mbase + q;
                        Vout[m][n][c] = (aV[t][q] + aT[t][q] * nav[m][c]) * cp;
                    }
                }
            }
        }
        __syncthreads();
        {
#pragma unroll
            for (int j = 0; j < 16; ++j) {
                int o = half * 16 + j;
                float s = Sbuf[r][o];
                Hs2[r][o] = s * sigm(s);
                float g = sigm(Sbuf[r][32 + o]);
#pragma unroll
                for (int c = 0; c < 3; ++c)
                    Vout[r][o][c] = g * Vout[r][o][c];
            }
        }
        __syncthreads();
        if (half == 0 && nd < N) {
#pragma unroll
            for (int o = 0; o < 2; ++o) {
                float t2 = 0.f;
#pragma unroll
                for (int i = 0; i < 32; ++i) t2 += w.sv2[o * 32 + i] * Hs2[r][i];
#pragma unroll
                for (int c = 0; c < 3; ++c) {
                    float v = 0.f;
#pragma unroll
                    for (int i = 0; i < 32; ++i) v += w.vs2[o * 32 + i] * Vout[r][i][c];
                    out[(size_t)nd * 6 + o * 3 + c] = (v + t2 * nav[r][c]) * cp;
                }
            }
        }
        __syncthreads();
    }
}

// ---------------------------------------------------------------------------
// Host side: param-tree walking (jax sorted-key flatten order), weight packing,
// layer scheduling. Everything on `stream` (graph-capture safe).
// ---------------------------------------------------------------------------
struct TPW { const float *ss, *sv, *vs, *vv, *b; };

struct Cursor {
    void* const* din;
    int idx;
    const float* base;
    size_t off;
    bool sep;
    const float* next(size_t n) {
        if (sep) return (const float*)din[idx++];
        const float* p = base + off;
        off += n;
        return p;
    }
};

// leaf order per tp dict (sorted keys): W_ss, W_sv, W_vs, W_vv, b
static TPW take_tp(Cursor& c, int o0, int m0, int o1, int m1) {
    TPW w;
    w.ss = c.next((size_t)o0 * m0);
    w.sv = c.next((size_t)o1 * m0);
    w.vs = c.next((size_t)o1 * m1);
    w.vv = c.next((size_t)o0 * m1);
    w.b  = c.next((size_t)o0);
    return w;
}

extern "C" void kernel_launch(void* const* d_in, const int* in_sizes, int n_in,
                              void* d_out, int out_size, void* d_ws, size_t ws_size,
                              hipStream_t stream) {
    const int N = in_sizes[0] / 7;
    const int E = in_sizes[1] / 2;
    const float* x    = (const float*)d_in[0];
    const int*   eidx = (const int*)d_in[1];
    const float* eatt = (const float*)d_in[2];
    const float* natt = (const float*)d_in[3];
    const float* amf  = (const float*)d_in[4];

    Cursor cur{ d_in, 5, (const float*)d_in[5], 0, n_in > 6 };
    TPW embed = take_tp(cur, 32, 1, 32, 2);
    TPW msg1[4], msg2[4], upd1[4], upd2[4];
    for (int l = 0; l < 4; ++l) {
        msg1[l] = take_tp(cur, 64, 66, 32, 64);
        msg2[l] = take_tp(cur, 64, 32, 32, 32);
        upd1[l] = take_tp(cur, 64, 64, 32, 64);
        upd2[l] = take_tp(cur, 32, 32, 32, 32);
    }
    TPW pre1 = take_tp(cur, 64, 32, 32, 32);
    TPW pre2 = take_tp(cur, 0, 32, 2, 32);

    char* wsb = (char*)d_ws;
    size_t off = 0;
    auto alloc = [&](size_t bytes) -> void* {
        size_t p = (off + 255) & ~(size_t)255;
        off = p + bytes;
        return (void*)(wsb + p);
    };
    float* hs   = (float*)alloc((size_t)N * 32 * 4);
    float* hv   = (float*)alloc((size_t)N * 96 * 4);
    float* ag   = (float*)alloc((size_t)N * 128 * 4);
    float* ag_s = ag;
    float* ag_v = ag + (size_t)N * 32;

    auto packB = [&](const float* W, int O, int K, int Kpad) -> const _Float16* {
        int ch = Kpad / 32, nt = (O + 15) / 16;
        _Float16* dst = (_Float16*)alloc((size_t)ch * nt * 512 * 2);
        pack_b_kernel<<<dim3(ch * nt), dim3(32), 0, stream>>>(W, O, K, nt, dst);
        return dst;
    };

    EdgeW ew[4];
    NodeW nw[4];
    for (int l = 0; l < 4; ++l) {
        ew[l].ss1 = packB(msg1[l].ss, 64, 66, 96);
        ew[l].vv1 = packB(msg1[l].vv, 64, 64, 64);
        ew[l].sv1 = packB(msg1[l].sv, 32, 66, 96);
        ew[l].vs1 = packB(msg1[l].vs, 32, 64, 64);
        ew[l].b1  = msg1[l].b;
        ew[l].ss2 = packB(msg2[l].ss, 64, 32, 32);
        ew[l].vv2 = packB(msg2[l].vv, 64, 32, 32);
        ew[l].sv2 = packB(msg2[l].sv, 32, 32, 32);
        ew[l].vs2 = packB(msg2[l].vs, 32, 32, 32);
        ew[l].b2  = msg2[l].b;
        nw[l].ss1 = packB(upd1[l].ss, 64, 64, 64);
        nw[l].vv1 = packB(upd1[l].vv, 64, 64, 64);
        nw[l].sv1 = packB(upd1[l].sv, 32, 64, 64);
        nw[l].vs1 = packB(upd1[l].vs, 32, 64, 64);
        nw[l].b1  = upd1[l].b;
        nw[l].ss2 = packB(upd2[l].ss, 32, 32, 32);
        nw[l].vv2 = packB(upd2[l].vv, 32, 32, 32);
        nw[l].sv2 = packB(upd2[l].sv, 32, 32, 32);
        nw[l].vs2 = packB(upd2[l].vs, 32, 32, 32);
        nw[l].b2  = upd2[l].b;
    }
    FinW fw;
    fw.ss1 = packB(pre1.ss, 64, 32, 32);
    fw.vv1 = packB(pre1.vv, 64, 32, 32);
    fw.sv1 = packB(pre1.sv, 32, 32, 32);
    fw.vs1 = packB(pre1.vs, 32, 32, 32);
    fw.b1  = pre1.b;
    fw.vs2 = pre2.vs;
    fw.sv2 = pre2.sv;

    (void)ws_size;
    (void)out_size;

    embed_kernel<<<(N + 127) / 128, 128, 0, stream>>>(
        x, natt, embed.ss, embed.sv, embed.vs, embed.vv, embed.b, hs, hv, N);

    const int ntE = (E + 15) / 16;
    const int ntN = (N + 15) / 16;
    for (int l = 0; l < 4; ++l) {
        hipMemsetAsync(ag, 0, (size_t)N * 128 * sizeof(float), stream);
        edge_msg_kernel<<<ntE, 32, 0, stream>>>(hs, hv, eidx, eatt, amf,
                                                ag_s, ag_v, ew[l], E);
        node_upd_kernel<<<ntN, 32, 0, stream>>>(hs, hv, ag_s, ag_v, natt, nw[l], N);
    }
    final_kernel<<<ntN, 32, 0, stream>>>(hs, hv, natt, fw, (float*)d_out, N);
}
